// SignAwareAttentionLayer_15109694947618
// MI455X (gfx1250) — matrix operand order repaired and checked by reading, not compiled, not run on hardware
//
#include <hip/hip_runtime.h>
#include <hip/hip_bf16.h>
#include <math.h>
#include <stdint.h>

#define N_NODES 100000
#define N_EDGES 1600000
#define DIM     64
#define KH      4
#define HD      16

typedef __attribute__((ext_vector_type(16))) _Float16 v16h;
typedef __attribute__((ext_vector_type(8)))  _Float16 v8h;
typedef __attribute__((ext_vector_type(4)))  _Float16 v4h;
typedef __attribute__((ext_vector_type(8)))  float    v8f;

union V16U { v16h v; v8h h8[2]; _Float16 e[16]; };
union V8FU { v8f v; float f[8]; };

// ---------------------------------------------------------------------------
// 64->64 GEMM: out[m][n] = sum_k in[m][k] * W[n][k] + bias[n]  (W row-major)
// MODE 0: plain   MODE 1: + (1+eps)*extra[m][n]   MODE 2: ReLU
// INHALF: input already f16 -> stage with global_load_async_to_lds_b128
// OUTHALF: store result as f16 (feeds the next WMMA GEMM)
// One wave computes a 16x64 row tile: 4 column tiles x (2 chained WMMAs, K=64).
// ---------------------------------------------------------------------------
template <int MODE, int INHALF, int OUTHALF>
__launch_bounds__(256, 2)
__global__ void gemm64_kernel(const void* __restrict__ inRaw,
                              const float* __restrict__ W,
                              const float* __restrict__ bias,
                              const float* __restrict__ extra,
                              const float* __restrict__ epsPtr,
                              void* __restrict__ outRaw, int n)
{
    __shared__ __align__(32) _Float16 sW[DIM * DIM];      // 8 KB
    __shared__ __align__(32) _Float16 sIn[8][16 * DIM];   // 16 KB (8 waves x 16 rows)
    __shared__ float sB[DIM];

    const int tid  = threadIdx.x;
    const int wave = tid >> 5;
    const int lane = tid & 31;
    const int rowBase   = (blockIdx.x * 8 + wave) * 16;
    const int blockRow0 = blockIdx.x * 128;

    // ---- stage weights (f32 -> f16, float4 vectorized) + bias ----
    for (int i = tid; i < DIM * 16; i += 256) {           // 1024 float4 chunks
        int r  = i >> 4;
        int c4 = (i & 15) * 4;
        float4 v = *(const float4*)(W + r * DIM + c4);
        v4h hv; hv[0] = (_Float16)v.x; hv[1] = (_Float16)v.y;
                hv[2] = (_Float16)v.z; hv[3] = (_Float16)v.w;
        *(v4h*)&sW[r * DIM + c4] = hv;
    }
    if (tid < DIM) sB[tid] = bias[tid];

    // ---- stage this block's 128 input rows ----
    if (INHALF) {
        // f16 input: async DMA straight into LDS, 16B per lane per issue.
        const _Float16* inh = (const _Float16*)inRaw;
        const unsigned ldsBase = (unsigned)(uintptr_t)&sIn[0][0];
        #pragma unroll
        for (int it = 0; it < 4; ++it) {
            int c    = tid + it * 256;        // 16B chunk id, 0..1023
            int r    = c >> 3;                // row in block (8 chunks per row)
            int colh = (c & 7) * 8;           // starting half within row
            unsigned lds = ldsBase +
                (unsigned)(((r >> 4) * 1024 + (r & 15) * DIM + colh) * 2);
            unsigned long long gaddr =
                (unsigned long long)(uintptr_t)(inh + (size_t)(blockRow0 + r) * DIM + colh);
            asm volatile("global_load_async_to_lds_b128 %0, %1, off"
                         :: "v"(lds), "v"(gaddr) : "memory");
        }
        asm volatile("s_wait_asynccnt 0x0" ::: "memory");
    } else {
        // f32 input: float4 load, convert, 8B LDS store.
        const float* in = (const float*)inRaw;
        for (int i = tid; i < 128 * 16; i += 256) {       // 2048 float4 chunks
            int r  = i >> 4;
            int c4 = (i & 15) * 4;
            int gr = blockRow0 + r;
            float4 v = make_float4(0.f, 0.f, 0.f, 0.f);
            if (gr < n) v = *(const float4*)(in + (size_t)gr * DIM + c4);
            v4h hv; hv[0] = (_Float16)v.x; hv[1] = (_Float16)v.y;
                    hv[2] = (_Float16)v.z; hv[3] = (_Float16)v.w;
            *(v4h*)&sIn[r >> 4][(r & 15) * DIM + c4] = hv;
        }
    }
    __syncthreads();

    // per-lane WMMA fragment geometry (ISA 7.12.2, 16-bit A 16x32 / B 32x16)
    const int mrow  = lane & 15;       // A row / B column / D column for this lane
    const int hi16  = lane >> 4;       // lane group
    const int koff  = hi16 * 8;        // unused (A and B both use khalf)
    const int khalf = hi16 * 16;       // B: K half offset
    const _Float16* arow = &sIn[wave][mrow * DIM];

    float eps0 = 0.0f;
    if (MODE == 1) eps0 = 1.0f + epsPtr[0];
    float*    outF = (float*)outRaw;
    _Float16* outH = (_Float16*)outRaw;

    // wave-uniform full-tile test -> branch-free store path for 781/782 blocks
    const int  rbu      = __builtin_amdgcn_readfirstlane(rowBase);
    const bool fullTile = (rbu + 16 <= n);

    #pragma unroll
    for (int nt = 0; nt < 4; ++nt) {
        const int ncol = nt * 16 + mrow;
        V8FU c;
        const float bval = sB[ncol];
        #pragma unroll
        for (int i = 0; i < 8; ++i) c.f[i] = bval;   // bias broadcast across rows

        #pragma unroll
        for (int kt = 0; kt < DIM; kt += 32) {
            V16U A, B;
            // A lane layout: e[j] = K kt+khalf+j (contiguous)
            A.h8[0] = *(const v8h*)(arow + kt + khalf);
            A.h8[1] = *(const v8h*)(arow + kt + khalf + 8);
            // B lane layout: e[j] = B[khalf+j][ncol] = W[ncol][kt+khalf+j] (contiguous)
            B.v = *(const v16h*)(&sW[ncol * DIM + kt + khalf]);
            c.v = __builtin_amdgcn_wmma_f32_16x16x32_f16(false, A.v, false, B.v,
                                                         (short)0, c.v, false, false);
        }

        // D layout: VGPR i holds row M = i + 8*hi16, column = ncol
        if (fullTile) {
            #pragma unroll
            for (int i = 0; i < 8; ++i) {
                int g = rowBase + i + hi16 * 8;
                float v = c.f[i];
                if (MODE == 1) v += eps0 * extra[(size_t)g * DIM + ncol];
                if (MODE == 2) v = v > 0.0f ? v : 0.0f;
                if (OUTHALF) outH[(size_t)g * DIM + ncol] = (_Float16)v;
                else         outF[(size_t)g * DIM + ncol] = v;
            }
        } else {
            #pragma unroll
            for (int i = 0; i < 8; ++i) {
                int g = rowBase + i + hi16 * 8;
                if (g < n) {
                    float v = c.f[i];
                    if (MODE == 1) v += eps0 * extra[(size_t)g * DIM + ncol];
                    if (MODE == 2) v = v > 0.0f ? v : 0.0f;
                    if (OUTHALF) outH[(size_t)g * DIM + ncol] = (_Float16)v;
                    else         outF[(size_t)g * DIM + ncol] = v;
                }
            }
        }
    }
}

// ---------------------------------------------------------------------------
// Edge phase helpers
// ---------------------------------------------------------------------------
__device__ __forceinline__ unsigned int f32_to_ord(float f) {
    unsigned int b = __float_as_uint(f);
    return (b & 0x80000000u) ? ~b : (b | 0x80000000u);
}
__device__ __forceinline__ float ord_to_f32(unsigned int u) {
    unsigned int b = (u & 0x80000000u) ? (u & 0x7FFFFFFFu) : ~u;
    return __uint_as_float(b);
}

__global__ void init_ws_kernel(float* __restrict__ aggr, float* __restrict__ den,
                               unsigned int* __restrict__ mxenc)
{
    int idx = blockIdx.x * 256 + threadIdx.x;
    if (idx < N_NODES * DIM) aggr[idx] = 0.0f;
    if (idx < N_NODES * KH) { den[idx] = 0.0f; mxenc[idx] = 0u; }
}

// one thread per (edge, head): score + ordered atomicMax into target segment
__global__ void score_kernel(const float* __restrict__ h,
                             const long long* __restrict__ ei,
                             const long long* __restrict__ esign,
                             const float* __restrict__ sign_tab,
                             float* __restrict__ score,
                             unsigned int* __restrict__ mxenc)
{
    int idx = blockIdx.x * 256 + threadIdx.x;
    if (idx >= N_EDGES * KH) return;
    int e = idx >> 2, k = idx & 3;
    int r = (int)ei[e];
    int c = (int)ei[N_EDGES + e];
    int s = (int)esign[e];
    const float4* hi = (const float4*)(h + (size_t)r * DIM + k * HD);
    const float4* hj = (const float4*)(h + (size_t)c * DIM + k * HD);
    const float4* st = (const float4*)(sign_tab + s * HD);
    float acc = 0.0f;
    #pragma unroll
    for (int q = 0; q < 4; ++q) {
        float4 a = hi[q], b = hj[q], t = st[q];
        acc += a.x * b.x * t.x + a.y * b.y * t.y + a.z * b.z * t.z + a.w * b.w * t.w;
    }
    score[idx] = acc;
    atomicMax(&mxenc[r * KH + k], f32_to_ord(acc));
}

// decode ordered-uint max -> float in place (enc==0 means empty segment -> 0)
__global__ void decode_max_kernel(unsigned int* __restrict__ mxenc,
                                  float* __restrict__ mx)
{
    int idx = blockIdx.x * 256 + threadIdx.x;
    if (idx >= N_NODES * KH) return;
    unsigned int u = mxenc[idx];
    mx[idx] = (u == 0u) ? 0.0f : ord_to_f32(u);
}

// ex = exp(score - mx[row]); den[row] += ex  (score buffer overwritten with ex)
__global__ void expsum_kernel(const long long* __restrict__ ei,
                              const float* __restrict__ mx,
                              float* __restrict__ score,
                              float* __restrict__ den)
{
    int idx = blockIdx.x * 256 + threadIdx.x;
    if (idx >= N_EDGES * KH) return;
    int e = idx >> 2, k = idx & 3;
    int r = (int)ei[e];
    float ex = expf(score[idx] - mx[r * KH + k]);
    score[idx] = ex;
    atomicAdd(&den[r * KH + k], ex);
}

// one thread per (edge, float4 chunk): aggr[row][d..d+3] += h[col][d..d+3]*alpha
__global__ void aggregate_kernel(const float* __restrict__ h,
                                 const long long* __restrict__ ei,
                                 const float* __restrict__ ex,
                                 const float* __restrict__ den,
                                 float* __restrict__ aggr)
{
    int idx = blockIdx.x * 256 + threadIdx.x;
    if (idx >= N_EDGES * 16) return;        // 25.6M threads
    int e = idx >> 4, q = idx & 15;         // q: which float4 of the 64 dims
    int k = q >> 2;                         // head = (q*4)/16
    int r = (int)ei[e];
    int c = (int)ei[N_EDGES + e];
    float alpha = ex[e * KH + k] / (den[r * KH + k] + 1e-16f);
    float4 hv = *(const float4*)(h + (size_t)c * DIM + q * 4);
    float* dst = aggr + (size_t)r * DIM + q * 4;
    atomicAdd(dst + 0, hv.x * alpha);
    atomicAdd(dst + 1, hv.y * alpha);
    atomicAdd(dst + 2, hv.z * alpha);
    atomicAdd(dst + 3, hv.w * alpha);
}

// ---------------------------------------------------------------------------
extern "C" void kernel_launch(void* const* d_in, const int* in_sizes, int n_in,
                              void* d_out, int out_size, void* d_ws, size_t ws_size,
                              hipStream_t stream)
{
    const float*     x        = (const float*)d_in[0];
    const long long* ei       = (const long long*)d_in[1];
    const long long* esign    = (const long long*)d_in[2];
    const float*     Wl_w     = (const float*)d_in[3];
    const float*     Wl_b     = (const float*)d_in[4];
    const float*     sign_tab = (const float*)d_in[5];
    const float*     Wh_w     = (const float*)d_in[6];
    const float*     Wh_b     = (const float*)d_in[7];
    const float*     w1       = (const float*)d_in[8];
    const float*     b1       = (const float*)d_in[9];
    const float*     w2       = (const float*)d_in[10];
    const float*     b2       = (const float*)d_in[11];
    const float*     eps      = (const float*)d_in[12];
    float* out = (float*)d_out;

    // workspace (floats): h | aggr | score(ex; later z16+hid16) | mx | den
    float* ws    = (float*)d_ws;
    float* h     = ws;                                  // N*64 f32
    float* aggr  = h     + (size_t)N_NODES * DIM;       // N*64 f32
    float* score = aggr  + (size_t)N_NODES * DIM;       // E*4  f32
    float* mx    = score + (size_t)N_EDGES * KH;        // N*4
    float* den   = mx    + (size_t)N_NODES * KH;        // N*4
    // after aggregation, score region is dead -> reuse for f16 intermediates
    _Float16* z16   = (_Float16*)score;                 // N*64 f16
    _Float16* hid16 = z16 + (size_t)N_NODES * DIM;      // N*64 f16

    const dim3 blk(256);
    const int gemmBlocks = (N_NODES + 127) / 128;

    // 1) h = x @ Wl^T + Wl_b                              (WMMA)
    gemm64_kernel<0,0,0><<<gemmBlocks, blk, 0, stream>>>(x, Wl_w, Wl_b, nullptr, nullptr, h, N_NODES);
    // 2) init aggregation / softmax state
    init_ws_kernel<<<(N_NODES * DIM + 255) / 256, blk, 0, stream>>>(aggr, den, (unsigned int*)mx);
    // 3) per-edge scores + segment max
    score_kernel<<<(N_EDGES * KH + 255) / 256, blk, 0, stream>>>(h, ei, esign, sign_tab, score, (unsigned int*)mx);
    // 4) decode segment max
    decode_max_kernel<<<(N_NODES * KH + 255) / 256, blk, 0, stream>>>((unsigned int*)mx, mx);
    // 5) exp + segment sum
    expsum_kernel<<<(N_EDGES * KH + 255) / 256, blk, 0, stream>>>(ei, mx, score, den);
    // 6) weighted scatter-add aggregation (L2-resident atomics)
    aggregate_kernel<<<(N_EDGES * 16 + 255) / 256, blk, 0, stream>>>(h, ei, score, den, aggr);
    // 7) z16 = f16( (1+eps)*h + aggr @ Wh^T + Wh_b )      (WMMA, fused epilogue)
    gemm64_kernel<1,0,1><<<gemmBlocks, blk, 0, stream>>>(aggr, Wh_w, Wh_b, h, eps, z16, N_NODES);
    // 8) hid16 = f16(relu(z @ w1^T + b1))                 (WMMA, async-LDS stage)
    gemm64_kernel<2,1,1><<<gemmBlocks, blk, 0, stream>>>(z16, w1, b1, nullptr, nullptr, hid16, N_NODES);
    // 9) out = hid @ w2^T + b2                            (WMMA, async-LDS stage)
    gemm64_kernel<0,1,0><<<gemmBlocks, blk, 0, stream>>>(hid16, w2, b2, nullptr, nullptr, out, N_NODES);
}
